// QLSTM_65481071398872
// MI455X (gfx1250) — matrix operand order repaired and checked by reading, not compiled, not run on hardware
//
#include <hip/hip_runtime.h>
#include <hip/hip_bf16.h>

// ---------------------------------------------------------------------------
// QLSTM for MI455X (gfx1250, wave32, WMMA bf16 16x16x32, f32 accumulate)
//   K1: input projection [S*B,256]x[256,1024] -> Gx, A staged via TDM
//       (tensor_load_to_lds with LDS padding), B streamed with global_load.
//   K2: persistent recurrence, 16 WGs, agent-scope fenced grid barrier,
//       Wh resident in registers as B-fragments for all 1024 steps.
//   K3a/3b: attention context + in-place finalize.
// ---------------------------------------------------------------------------

#define S_LEN 1024
#define B_SZ  64
#define D_IN  256
#define H_SZ  256
#define KW    512          // D+H, W row stride
#define NCOL  1024         // 4*H gate columns

typedef __bf16 bf16;
typedef __attribute__((ext_vector_type(16))) __bf16 v16bf;
typedef __attribute__((ext_vector_type(2)))  __bf16 v2bf;
typedef __attribute__((ext_vector_type(8)))  float  v8f;
typedef __attribute__((ext_vector_type(4)))  unsigned int u32x4;
typedef __attribute__((ext_vector_type(8)))  int    i32x8;
typedef __attribute__((ext_vector_type(4)))  int    i32x4;

__device__ __forceinline__ float sigmoid_f(float x) {
    return 1.0f / (1.0f + __expf(-x));
}

// ---------------------------------------------------------------------------
// Kernel 0: zero h_0 ping buffer and grid-barrier counter.
// ---------------------------------------------------------------------------
__global__ void qlstm_init(float* __restrict__ hbuf, int* __restrict__ counter) {
    int i = blockIdx.x * 256 + threadIdx.x;
    if (i < B_SZ * H_SZ) hbuf[i] = 0.0f;
    if (i == 0) *counter = 0;
}

// ---------------------------------------------------------------------------
// Kernel 1: Gx[s][b][g][h] = sum_k x[s][b][k]*W_g[h][k] + b_g[h] + th_g[h]
// 256 thr = 8 waves, each wave owns one 16-row M tile of S*B.
// A (x) is staged into LDS by the Tensor Data Mover in 4 K-quarters:
//   2D tile 64 cols x 128 rows fp32, tensor stride 256, TDM pad 4 DW per
//   64 DW -> LDS row stride 68 floats (bank-conflict-free fragment gather).
// ---------------------------------------------------------------------------
__global__ __launch_bounds__(256) void qlstm_xproj(
    const float* __restrict__ x,
    const float* __restrict__ Wf, const float* __restrict__ Wi,
    const float* __restrict__ Wu, const float* __restrict__ Wo,
    const float* __restrict__ bf_, const float* __restrict__ bi_,
    const float* __restrict__ bu_, const float* __restrict__ bo_,
    const float* __restrict__ thf, const float* __restrict__ thi,
    const float* __restrict__ thu, const float* __restrict__ tho,
    float* __restrict__ Gx)
{
    __shared__ float Alds[128][68];          // 64-col K quarter, padded (34 KB)

    const int wave  = threadIdx.x >> 5;
    const int lane  = threadIdx.x & 31;
    const int lan16 = lane & 15;
    const int hi16  = lane >> 4;             // 0 or 1
    const int mblock = blockIdx.x * 128;     // row offset in [S*B]

    v16bf afrag[8];                          // 16x32 K tiles covering K=256
    const int m   = wave * 16 + lan16;
    const int kh8 = hi16 << 3;

#pragma unroll
    for (int q = 0; q < 4; ++q) {            // K quarters of 64
#if __has_builtin(__builtin_amdgcn_tensor_load_to_lds)
        if (threadIdx.x < 32) {
            const float* gp = x + (size_t)mblock * D_IN + q * 64;
            unsigned long long ga = (unsigned long long)(size_t)gp;
            unsigned lds_base = (unsigned)(size_t)&Alds[0][0];
            u32x4 g0 = {1u,                              // count=1 (valid D#)
                        lds_base,                        // lds_addr
                        (unsigned)(ga & 0xFFFFFFFFu),
                        (unsigned)((ga >> 32) & 0x1FFFFFFu) | (2u << 30)}; // type=2
            // data_size=4B, pad_enable, pad_interval=64DW(code5), pad_amount=4DW(code3)
            i32x8 g1 = {(2 << 16) | (1 << 20) | (5 << 22) | (3 << 25),
                        0,                               // atomic barrier addr
                        0x4000,                          // tensor_dim0 = 1<<30 (hi16)
                        (64 << 16) | 0x10,               // tile_dim0=64 | tensor_dim1=1<<20 (hi16)
                        128,                             // tile_dim1=128, tile_dim2=0
                        256,                             // tensor_dim0_stride lo
                        0, 0};                           // stride hi, dim1_stride
            i32x4 g2 = {0, 0, 0, 0};
            i32x4 g3 = {0, 0, 0, 0};
            i32x8 g4 = {0, 0, 0, 0, 0, 0, 0, 0};
            __builtin_amdgcn_tensor_load_to_lds(g0, g1, g2, g3, g4, 0);
            __builtin_amdgcn_s_wait_tensorcnt(0);
        }
        __syncthreads();
#else
        for (int i = threadIdx.x; i < 128 * 64; i += 256) {
            int r = i >> 6, k = i & 63;
            Alds[r][k] = x[(size_t)(mblock + r) * D_IN + q * 64 + k];
        }
        __syncthreads();
#endif
        // build the 2 A-fragments of this quarter (ISA 16-bit A layout)
#pragma unroll
        for (int ktl = 0; ktl < 2; ++ktl) {
#pragma unroll
            for (int v = 0; v < 8; ++v) {
                int kq = ktl * 32 + (v & 3) * 2 + kh8 + ((v >> 2) << 4);
                afrag[q * 2 + ktl][2 * v]     = (bf16)Alds[m][kq];
                afrag[q * 2 + ktl][2 * v + 1] = (bf16)Alds[m][kq + 1];
            }
        }
        __syncthreads();                     // before TDM overwrites quarter
    }

    const int mrow_base = mblock + wave * 16 + (hi16 << 3);
    const int koff = hi16 << 3;              // B frag: lanes 16-31 hold K+8 (table order)

#pragma unroll
    for (int g = 0; g < 4; ++g) {            // unrolled: direct kernarg ptrs
        const float* W  = (g == 0) ? Wf : (g == 1) ? Wi : (g == 2) ? Wu : Wo;
        const float* bp = (g == 0) ? bf_ : (g == 1) ? bi_ : (g == 2) ? bu_ : bo_;
        const float* tp = (g == 0) ? thf : (g == 1) ? thi : (g == 2) ? thu : tho;
        for (int nt = 0; nt < 16; ++nt) {
            const int h = nt * 16 + lan16;
            const float bias = bp[h] + tp[h];

            v8f acc;
#pragma unroll
            for (int v = 0; v < 8; ++v) acc[v] = bias;   // fold b+th into C

            const float* wrow = W + (size_t)h * KW;      // x-part cols 0..255
#pragma unroll
            for (int kt = 0; kt < 8; ++kt) {
                v16bf bfrag;
                const float* p = wrow + kt * 32 + koff;
#pragma unroll
                for (int e = 0; e < 16; ++e) bfrag[e] = (bf16)p[(e & 7) + ((e >> 3) << 4)];
                acc = __builtin_amdgcn_wmma_f32_16x16x32_bf16(
                    false, afrag[kt], false, bfrag, (short)0, acc, false, false);
            }
            const int col = g * H_SZ + nt * 16 + lan16;
#pragma unroll
            for (int v = 0; v < 8; ++v)
                Gx[(size_t)(mrow_base + v) * NCOL + col] = acc[v];
        }
    }
}

// ---------------------------------------------------------------------------
// Kernel 2: persistent recurrence.  16 WGs x 256 threads (8 waves).
// WG wg owns hidden units [wg*16, wg*16+16) across all 4 gates.
// Wave w: gate g=w&3, M tiles {2*(w>>2), 2*(w>>2)+1}.  Wh B-fragments in
// registers for all steps.  h_{t-1} staged as packed bf16 pairs in LDS.
// ---------------------------------------------------------------------------
__global__ __launch_bounds__(256) void qlstm_recur(
    const float* __restrict__ Wf, const float* __restrict__ Wi,
    const float* __restrict__ Wu, const float* __restrict__ Wo,
    const float* __restrict__ Gx,
    float* __restrict__ hbuf,                // [2][B][H] ping-pong
    float* __restrict__ out,                 // [B][S][H] raw h storage
    float* __restrict__ hx, float* __restrict__ cx,
    int* __restrict__ counter)
{
    __shared__ unsigned hlds[B_SZ][130];     // bf16 pairs, padded (33 KB)
    __shared__ float pre[4][B_SZ][16];       // gate pre-activations (16 KB)
    __shared__ float cst[B_SZ][16];          // cell state slice (4 KB)

    const int wg    = blockIdx.x;            // 0..15
    const int wave  = threadIdx.x >> 5;
    const int lane  = threadIdx.x & 31;
    const int lan16 = lane & 15;
    const int hi16  = lane >> 4;
    const int g     = wave & 3;
    const int mt0   = (wave >> 2) * 2;

    // Wh slice once, as ready-to-use B fragments (ternary keeps AS inference)
    const float* Wp = (g == 0) ? Wf : (g == 1) ? Wi : (g == 2) ? Wu : Wo;
    v16bf wfrag[8];
    {
        const int h = wg * 16 + lan16;
        const float* wrow = Wp + (size_t)h * KW + D_IN;   // hidden cols
        const int koff = hi16 << 3;
#pragma unroll
        for (int kt = 0; kt < 8; ++kt) {
            const float* p = wrow + kt * 32 + koff;
#pragma unroll
            for (int e = 0; e < 16; ++e) wfrag[kt][e] = (bf16)p[(e & 7) + ((e >> 3) << 4)];
        }
    }
    for (int i = threadIdx.x; i < B_SZ * 16; i += 256) cst[i >> 4][i & 15] = 0.0f;
    __syncthreads();

    for (int s = 0; s < S_LEN; ++s) {
        const float* hprev = hbuf + (size_t)(s & 1) * (B_SZ * H_SZ);
        float*       hnext = hbuf + (size_t)((s + 1) & 1) * (B_SZ * H_SZ);

        // stage h_{t-1} -> LDS as packed bf16 pairs (coalesced)
        for (int i = threadIdx.x; i < B_SZ * H_SZ / 2; i += 256) {
            v2bf pk;
            pk[0] = (bf16)hprev[2 * i];
            pk[1] = (bf16)hprev[2 * i + 1];
            hlds[i >> 7][i & 127] = __builtin_bit_cast(unsigned, pk);
        }
        __syncthreads();

#pragma unroll
        for (int t = 0; t < 2; ++t) {
            const int mt = mt0 + t;
            const int brow = mt * 16 + (hi16 << 3);
            const int col  = g * H_SZ + wg * 16 + lan16;

            v8f acc;                                     // C := Gx (has b+th)
#pragma unroll
            for (int v = 0; v < 8; ++v)
                acc[v] = Gx[(size_t)(s * B_SZ + brow + v) * NCOL + col];

            const int m = mt * 16 + lan16;
            const int kh8 = hi16 << 3;
#pragma unroll
            for (int kt = 0; kt < 8; ++kt) {
                v16bf a;
#pragma unroll
                for (int v = 0; v < 8; ++v) {
                    int kq = kt * 32 + (v & 3) * 2 + kh8 + ((v >> 2) << 4);
                    v2bf p2 = __builtin_bit_cast(v2bf, hlds[m][kq >> 1]);
                    a[2 * v]     = p2[0];
                    a[2 * v + 1] = p2[1];
                }
                acc = __builtin_amdgcn_wmma_f32_16x16x32_bf16(
                    false, a, false, wfrag[kt], (short)0, acc, false, false);
            }
#pragma unroll
            for (int v = 0; v < 8; ++v) pre[g][brow + v][lan16] = acc[v];
        }
        __syncthreads();

        // gate math + state update for this WG's 64x16 slice
        for (int i = threadIdx.x; i < B_SZ * 16; i += 256) {
            const int b = i >> 4, n = i & 15;
            const float fv = sigmoid_f(__cosf(pre[0][b][n]));
            const float iv = sigmoid_f(__cosf(pre[1][b][n]));
            const float uv = tanhf(__cosf(pre[2][b][n]));
            const float ov = sigmoid_f(__cosf(pre[3][b][n]));
            const float c  = fv * cst[b][n] + iv * uv;
            cst[b][n] = c;
            const float h = ov * tanhf(c);
            const int hg = wg * 16 + n;
            hnext[b * H_SZ + hg] = h;
            out[((size_t)b * S_LEN + s) * H_SZ + hg] = h;
            if (s == S_LEN - 1) hx[b * H_SZ + hg] = h;
        }

        // grid-wide step barrier with per-wave release/acquire at agent scope
        __builtin_amdgcn_fence(__ATOMIC_RELEASE, "agent");  // drain own stores
        __syncthreads();
        if (threadIdx.x == 0) {
            __hip_atomic_fetch_add(counter, 1, __ATOMIC_ACQ_REL,
                                   __HIP_MEMORY_SCOPE_AGENT);
            while (__hip_atomic_load(counter, __ATOMIC_ACQUIRE,
                                     __HIP_MEMORY_SCOPE_AGENT) < 16 * (s + 1))
                __builtin_amdgcn_s_sleep(1);
        }
        __syncthreads();
        __builtin_amdgcn_fence(__ATOMIC_ACQUIRE, "agent"); // invalidate L0s
    }

    for (int i = threadIdx.x; i < B_SZ * 16; i += 256) {
        const int b = i >> 4, n = i & 15;
        cx[b * H_SZ + wg * 16 + n] = cst[b][n];
    }
}

// ---------------------------------------------------------------------------
// Kernel 3a: attention context per (b,h).  tanh output bounded -> softmax
// needs no max shift: ctx = sum(e^t * y) / sum(e^t).  One block per batch
// row -> fully coalesced 1 KB reads per step.
// ---------------------------------------------------------------------------
__global__ __launch_bounds__(256) void qlstm_ctx(
    const float* __restrict__ out, const float* __restrict__ phi,
    const float* __restrict__ tha, float* __restrict__ ctxb)
{
    const int b = blockIdx.x;
    const int h = threadIdx.x;
    const float ph = phi[h], ta = tha[h];
    float se = 0.0f, sy = 0.0f;
    const float* p = out + (size_t)b * S_LEN * H_SZ + h;
    for (int s = 0; s < S_LEN; ++s) {
        const float y = __cosf(p[(size_t)s * H_SZ] + ph);
        const float t = tanhf(__cosf(y + ta));
        const float e = __expf(t);
        se += e;
        sy += e * y;
    }
    ctxb[b * H_SZ + h] = sy / se;
}

// ---------------------------------------------------------------------------
// Kernel 3b: in-place finalize  out = cos(raw + phi) + ctx[b][h]
// ---------------------------------------------------------------------------
__global__ __launch_bounds__(256) void qlstm_final(
    float* __restrict__ out, const float* __restrict__ phi,
    const float* __restrict__ ctxb)
{
    const size_t idx = (size_t)blockIdx.x * 256 + threadIdx.x;
    const int h = (int)(idx & 255);
    const int b = (int)(idx >> 18);          // idx = b*S*H + s*H + h
    out[idx] = __cosf(out[idx] + phi[h]) + ctxb[b * H_SZ + h];
}

// ---------------------------------------------------------------------------
extern "C" void kernel_launch(void* const* d_in, const int* in_sizes, int n_in,
                              void* d_out, int out_size, void* d_ws, size_t ws_size,
                              hipStream_t stream)
{
    (void)in_sizes; (void)n_in; (void)out_size; (void)ws_size;

    const float* x   = (const float*)d_in[0];
    const float* Wf  = (const float*)d_in[1];
    const float* bf_ = (const float*)d_in[2];
    const float* Wi  = (const float*)d_in[3];
    const float* bi_ = (const float*)d_in[4];
    const float* Wu  = (const float*)d_in[5];
    const float* bu_ = (const float*)d_in[6];
    const float* Wo  = (const float*)d_in[7];
    const float* bo_ = (const float*)d_in[8];
    const float* thf = (const float*)d_in[9];
    const float* thi = (const float*)d_in[10];
    const float* thu = (const float*)d_in[11];
    const float* tho = (const float*)d_in[12];
    const float* phi = (const float*)d_in[13];
    const float* tha = (const float*)d_in[14];

    float* out = (float*)d_out;                              // [B][S][H]
    float* hx  = out + (size_t)B_SZ * S_LEN * H_SZ;          // [B][H]
    float* cx  = hx + (size_t)B_SZ * H_SZ;                   // [B][H]

    // workspace: Gx (268.4 MB) | hbuf ping-pong | ctx | barrier counter
    float* Gx   = (float*)d_ws;
    float* hbuf = Gx + (size_t)S_LEN * B_SZ * NCOL;
    float* ctxb = hbuf + 2 * (size_t)B_SZ * H_SZ;
    int*   cnt  = (int*)(ctxb + (size_t)B_SZ * H_SZ);

    qlstm_init<<<(B_SZ * H_SZ + 255) / 256, 256, 0, stream>>>(hbuf, cnt);
    qlstm_xproj<<<(S_LEN * B_SZ) / 128, 256, 0, stream>>>(
        x, Wf, Wi, Wu, Wo, bf_, bi_, bu_, bo_, thf, thi, thu, tho, Gx);
    qlstm_recur<<<16, 256, 0, stream>>>(
        Wf, Wi, Wu, Wo, Gx, hbuf, out, hx, cx, cnt);
    qlstm_ctx<<<B_SZ, 256, 0, stream>>>(out, phi, tha, ctxb);
    qlstm_final<<<(int)(((size_t)B_SZ * S_LEN * H_SZ) / 256), 256, 0, stream>>>(
        out, phi, ctxb);
}
